// Transformer_28338194219350
// MI455X (gfx1250) — matrix operand order repaired and checked by reading, not compiled, not run on hardware
//
#include <hip/hip_runtime.h>
#include <hip/hip_bf16.h>

typedef __attribute__((ext_vector_type(16))) __bf16 v16bf;
typedef __attribute__((ext_vector_type(2)))  __bf16 v2bf;
typedef __attribute__((ext_vector_type(8)))  float  v8f;

struct __align__(16) U4 { unsigned int x, y, z, w; };
union FragAB { v16bf v; U4 q[2]; };

constexpr int Bb = 8192, Nn = 4096, Pp = 8, Dd = 512;
constexpr int BM = 128, BN = 128, BK = 32;
constexpr int KITERS = Dd / BK;   // 16
constexpr int AROW = 5;           // U4s per LDS row: 80B stride -> conflict-free b128
constexpr int BROW = 5;
constexpr float EPSF = 1e-5f;

__device__ inline unsigned short f2bf(float f) {
  union { __bf16 b; unsigned short u; } t;
  t.b = (__bf16)f;
  return t.u;
}
__device__ inline unsigned int pk2(float a, float b) {
  union { v2bf v; unsigned int u; } t;
  t.v[0] = (__bf16)a;
  t.v[1] = (__bf16)b;
  return t.u;
}

// ---- CDNA5 async global->LDS copy (ASYNCcnt) ----
__device__ inline void async_copy16(const void* g, void* l) {
  unsigned lofs = (unsigned)(size_t)l;            // flat shared addr[31:0] = LDS offset
  unsigned long long ga = (unsigned long long)(size_t)g;
  asm volatile("global_load_async_to_lds_b128 %0, %1, off"
               :: "v"(lofs), "v"(ga) : "memory");
}
__device__ inline void wait_async() {
  asm volatile("s_wait_asynccnt 0x0" ::: "memory");
}

// ---------------------------------------------------------------------------
// Block-diagonal GEMM, WMMA bf16 / f32 accumulate, double-buffered pipeline.
// PHASE 1: h = x @ W1t + b1 ; emit h (bf16) + per-column sum/sumsq atomics
// PHASE 2: o3 = h @ W2t' + b2' + x ; gated-sigmoid epilogue -> out (f32)
// 256 threads = 8 waves (4Mx2N); tile 128x128; wave: 32Mx64N; K-step 32.
// ---------------------------------------------------------------------------
template <int PHASE>
__global__ __launch_bounds__(256) void gemm_bd(
    const float* __restrict__ xf,
    const unsigned short* __restrict__ hin,
    const unsigned short* __restrict__ wt,    // bf16 [p][e][k]
    const float* __restrict__ bias,
    unsigned short* __restrict__ hout,
    float* __restrict__ colsum,
    float* __restrict__ colsumsq,
    const float* __restrict__ gamma,
    const float* __restrict__ beta,
    float* __restrict__ out) {
  __shared__ U4 ldsA[2][BM * AROW];
  __shared__ U4 ldsB[2][BN * BROW];

  const int tile_m = blockIdx.x * BM;
  const int ct     = blockIdx.y;              // 0..31: p*4 + col-tile
  const int p      = ct >> 2;
  const int colb   = (ct & 3) * BN;
  const int col0   = p * Dd + colb;

  const int tid    = threadIdx.x;
  const int wave   = tid >> 5;
  const int lane   = tid & 31;
  const int lane16 = lane & 15;
  const int half   = lane >> 4;
  const int wrow   = (wave >> 1) * 32;        // 0,32,64,96
  const int wcol   = (wave & 1) * 64;         // 0,64

  const int srow  = tid >> 1;      // A staging: row 0..127
  const int shalf = tid & 1;       // A staging: K-half
  const int brow  = tid >> 1;      // B staging: row 0..127
  const int bq2   = (tid & 1) * 2; // B staging: 32B chunk pair

  const unsigned short* browp = wt + ((size_t)p * Dd + colb + brow) * Dd + bq2 * 8;
  const size_t arow_off = (size_t)(tile_m + srow) * Nn + p * Dd + shalf * 16;

  v8f acc[2][4] = {};

  // ---- prologue: stage tile 0 into buffer 0 ----
  async_copy16(browp,     &ldsB[0][brow * BROW + bq2 + 0]);
  async_copy16(browp + 8, &ldsB[0][brow * BROW + bq2 + 1]);
  if constexpr (PHASE == 1) {
    const float4* f4 = (const float4*)(xf + arow_off);
    float4 v0 = f4[0], v1 = f4[1], v2 = f4[2], v3 = f4[3];
    U4 o0 = { pk2(v0.x, v0.y), pk2(v0.z, v0.w), pk2(v1.x, v1.y), pk2(v1.z, v1.w) };
    U4 o1 = { pk2(v2.x, v2.y), pk2(v2.z, v2.w), pk2(v3.x, v3.y), pk2(v3.z, v3.w) };
    ldsA[0][srow * AROW + 2 * shalf + 0] = o0;
    ldsA[0][srow * AROW + 2 * shalf + 1] = o1;
  } else {
    const unsigned short* src = hin + arow_off;
    async_copy16(src,     &ldsA[0][srow * AROW + 2 * shalf + 0]);
    async_copy16(src + 8, &ldsA[0][srow * AROW + 2 * shalf + 1]);
  }

  for (int it = 0; it < KITERS; ++it) {
    wait_async();          // tile `it` async arrivals done (issued one compute-phase ago)
    __syncthreads();       // all waves' stores for tile `it` visible; frees other buffer

    const int cur = it & 1;
    const int nxt = cur ^ 1;
    const bool have_next = (it + 1 < KITERS);
    const int k1 = (it + 1) * BK;

    // ---- issue stage of tile it+1 (overlaps with WMMA below) ----
    float4 v0{0.f,0.f,0.f,0.f}, v1{0.f,0.f,0.f,0.f}, v2{0.f,0.f,0.f,0.f}, v3{0.f,0.f,0.f,0.f};
    if (have_next) {
      async_copy16(browp + k1,     &ldsB[nxt][brow * BROW + bq2 + 0]);
      async_copy16(browp + k1 + 8, &ldsB[nxt][brow * BROW + bq2 + 1]);
      if constexpr (PHASE == 1) {
        const float4* f4 = (const float4*)(xf + arow_off + k1);
        v0 = f4[0]; v1 = f4[1]; v2 = f4[2]; v3 = f4[3];   // global loads in flight
      } else {
        const unsigned short* src = hin + arow_off + k1;
        async_copy16(src,     &ldsA[nxt][srow * AROW + 2 * shalf + 0]);
        async_copy16(src + 8, &ldsA[nxt][srow * AROW + 2 * shalf + 1]);
      }
    }

    // ---- compute on tile `it`: 2 A frags x 4 B frags = 8 WMMA ----
    FragAB a[2];
#pragma unroll
    for (int ai = 0; ai < 2; ++ai) {
      const U4* arow = &ldsA[cur][(wrow + ai * 16 + lane16) * AROW];
      a[ai].q[0] = arow[half];
      a[ai].q[1] = arow[2 + half];
    }
#pragma unroll
    for (int j = 0; j < 4; ++j) {
      const U4* bp = &ldsB[cur][(wcol + j * 16 + lane16) * BROW];
      FragAB b;
      b.q[0] = bp[half];           // K = 8*half .. +7
      b.q[1] = bp[2 + half];       // K = 16+8*half .. +7
#pragma unroll
      for (int ai = 0; ai < 2; ++ai) {
        acc[ai][j] = __builtin_amdgcn_wmma_f32_16x16x32_bf16(
            false, a[ai].v, false, b.v, (short)0, acc[ai][j], false, false);
      }
    }

    // ---- commit phase-1 A tile it+1 (after fragment ds_loads) ----
    if constexpr (PHASE == 1) {
      if (have_next) {
        U4 o0 = { pk2(v0.x, v0.y), pk2(v0.z, v0.w), pk2(v1.x, v1.y), pk2(v1.z, v1.w) };
        U4 o1 = { pk2(v2.x, v2.y), pk2(v2.z, v2.w), pk2(v3.x, v3.y), pk2(v3.z, v3.w) };
        ldsA[nxt][srow * AROW + 2 * shalf + 0] = o0;
        ldsA[nxt][srow * AROW + 2 * shalf + 1] = o1;
      }
    }
  }

  // ---- epilogue (C layout: lane=N, VGPR r -> M = r + 8*half) ----
#pragma unroll
  for (int ai = 0; ai < 2; ++ai) {
    const int m0 = tile_m + wrow + ai * 16 + half * 8;
#pragma unroll
    for (int j = 0; j < 4; ++j) {
      const int col = col0 + wcol + j * 16 + lane16;
      const float bv = bias[col];
      if constexpr (PHASE == 1) {
        float s1 = 0.f, s2 = 0.f;
#pragma unroll
        for (int r = 0; r < 8; ++r) {
          float h = acc[ai][j][r] + bv;
          s1 += h; s2 += h * h;
          hout[(size_t)(m0 + r) * Nn + col] = f2bf(h);
        }
        atomicAdd(&colsum[col], s1);
        atomicAdd(&colsumsq[col], s2);
      } else {
        const float g  = gamma[col];
        const float be = beta[col];
#pragma unroll
        for (int r = 0; r < 8; ++r) {
          float o3 = acc[ai][j][r] + bv + xf[(size_t)(m0 + r) * Nn + col];
          float sg = 1.0f / (1.0f + __expf(-be * o3));
          out[(size_t)(m0 + r) * Nn + col] = (g + sg * (1.0f - g)) * o3;
        }
      }
    }
  }
}

__global__ void zero_kernel(float* __restrict__ ptr, int n) {
  int i = blockIdx.x * blockDim.x + threadIdx.x;
  if (i < n) ptr[i] = 0.0f;
}

__global__ void stats_kernel(const float* __restrict__ colsum,
                             const float* __restrict__ colsumsq,
                             const float* __restrict__ bnw,
                             const float* __restrict__ bnb,
                             float* __restrict__ aArr, float* __restrict__ cArr) {
  int i = blockIdx.x * blockDim.x + threadIdx.x;
  if (i < Nn) {
    const float invB = 1.0f / (float)Bb;
    float mean = colsum[i] * invB;
    float var  = colsumsq[i] * invB - mean * mean;
    float a    = bnw[i] * rsqrtf(var + EPSF);
    aArr[i] = a;
    cArr[i] = bnb[i] - mean * a;
  }
}

__global__ void fold_bias_kernel(const float* __restrict__ cArr,
                                 const float* __restrict__ w2,
                                 const float* __restrict__ bias2,
                                 float* __restrict__ bias2p) {
  int i = blockIdx.x * blockDim.x + threadIdx.x;
  if (i >= Nn) return;
  int p = i >> 9, e = i & (Dd - 1);
  const float* wp = w2 + (size_t)p * Dd * Dd + e;
  const float* cp = cArr + p * Dd;
  float acc = 0.f;
  for (int k = 0; k < Dd; ++k) acc = fmaf(cp[k], wp[(size_t)k * Dd], acc);
  bias2p[i] = bias2[i] + acc;
}

__global__ __launch_bounds__(256) void transpose_scale_kernel(
    const float* __restrict__ src, const float* __restrict__ scale,
    unsigned short* __restrict__ dst) {
  __shared__ __align__(16) unsigned short tr[64][72];
  const int bid = blockIdx.x;
  const int p  = bid >> 6;
  const int kb = (bid >> 3) & 7;
  const int eb = bid & 7;
  const int t = threadIdx.x;
  {
    const int kl = t >> 2;
    const int eg = t & 3;
    const float s = scale ? scale[p * Dd + kb * 64 + kl] : 1.0f;
    const float* srow = src + ((size_t)p * Dd + kb * 64 + kl) * Dd + eb * 64 + eg * 16;
    const float4* f4 = (const float4*)srow;
#pragma unroll
    for (int i = 0; i < 4; ++i) {
      float4 v = f4[i];
      int e0 = eg * 16 + i * 4;
      tr[e0 + 0][kl] = f2bf(v.x * s);
      tr[e0 + 1][kl] = f2bf(v.y * s);
      tr[e0 + 2][kl] = f2bf(v.z * s);
      tr[e0 + 3][kl] = f2bf(v.w * s);
    }
  }
  __syncthreads();
  {
    const int el = t >> 2;
    const int kg = t & 3;
    const U4* row = (const U4*)&tr[el][0];
    U4 q0 = row[kg * 2 + 0];
    U4 q1 = row[kg * 2 + 1];
    unsigned short* dp = dst + ((size_t)p * Dd + eb * 64 + el) * Dd + kb * 64 + kg * 16;
    ((U4*)dp)[0] = q0;
    ((U4*)dp)[1] = q1;
  }
}

extern "C" void kernel_launch(void* const* d_in, const int* in_sizes, int n_in,
                              void* d_out, int out_size, void* d_ws, size_t ws_size,
                              hipStream_t stream) {
  const float* x   = (const float*)d_in[0];
  const float* w1  = (const float*)d_in[1];
  const float* b1  = (const float*)d_in[2];
  const float* w2  = (const float*)d_in[3];
  const float* b2  = (const float*)d_in[4];
  const float* bnw = (const float*)d_in[5];
  const float* bnb = (const float*)d_in[6];
  const float* g3  = (const float*)d_in[7];
  const float* be3 = (const float*)d_in[8];
  float* out = (float*)d_out;

  char* w = (char*)d_ws;
  size_t off = 0;
  unsigned short* hbf = (unsigned short*)(w + off); off += (size_t)Bb * Nn * 2;
  unsigned short* wt1 = (unsigned short*)(w + off); off += (size_t)Pp * Dd * Dd * 2;
  unsigned short* wt2 = (unsigned short*)(w + off); off += (size_t)Pp * Dd * Dd * 2;
  float* colsum   = (float*)(w + off); off += (size_t)Nn * 4;
  float* colsumsq = (float*)(w + off); off += (size_t)Nn * 4;
  float* aArr     = (float*)(w + off); off += (size_t)Nn * 4;
  float* cArr     = (float*)(w + off); off += (size_t)Nn * 4;
  float* bias2p   = (float*)(w + off); off += (size_t)Nn * 4;

  zero_kernel<<<(2 * Nn + 255) / 256, 256, 0, stream>>>(colsum, 2 * Nn);
  transpose_scale_kernel<<<Pp * 64, 256, 0, stream>>>(w1, nullptr, wt1);

  dim3 gg(Bb / BM, (Pp * Dd) / BN);   // (64, 32)
  gemm_bd<1><<<gg, 256, 0, stream>>>(x, nullptr, wt1, b1, hbf, colsum, colsumsq,
                                     nullptr, nullptr, nullptr);
  stats_kernel<<<(Nn + 255) / 256, 256, 0, stream>>>(colsum, colsumsq, bnw, bnb,
                                                     aArr, cArr);
  fold_bias_kernel<<<(Nn + 255) / 256, 256, 0, stream>>>(cArr, w2, b2, bias2p);
  transpose_scale_kernel<<<Pp * 64, 256, 0, stream>>>(w2, aArr, wt2);
  gemm_bd<2><<<gg, 256, 0, stream>>>(x, hbf, wt2, bias2p, nullptr, nullptr, nullptr,
                                     g3, be3, out);
}
